// SAGELayer_5403068859078
// MI455X (gfx1250) — compile-verified, matrix operand order checked by reading, not run on hardware
//
#include <hip/hip_runtime.h>

typedef _Float16 f16;
typedef __attribute__((ext_vector_type(16))) _Float16 v16h;
typedef __attribute__((ext_vector_type(8)))  _Float16 v8h;
typedef __attribute__((ext_vector_type(4)))  _Float16 v4h;
typedef __attribute__((ext_vector_type(2)))  _Float16 v2h;
typedef __attribute__((ext_vector_type(8)))  float    v8f;

#define N_NODES 50000
#define NNB     16
#define DIN     128
#define DE      64
#define DOUT    128
#define KMSG    (DIN + DE)     // 192
#define KAPP    (DIN + DOUT)   // 256
#define S_MSG   (KMSG / 32)    // 6 K-steps for GEMM1
#define S_APP   (KAPP / 32)    // 8 K-steps for GEMM2
#define NT      (DOUT / 16)    // 8 output-channel tiles

// ---------------------------------------------------------------------------
// Kernel 1: repack f32 weights into f16 WMMA B-fragment order.
// Fragment linear index = (t*steps + s)*32 + lane, 16 halfs each (32B).
// Per ISA 16-bit operand layout: lane<16 holds K{0..7,16..23} of the 32-step;
// lane>=16 holds K{8..15,24..31}.  B column n = 16*t + (lane&15).
// ---------------------------------------------------------------------------
__global__ __launch_bounds__(256) void repack_weights(const float* __restrict__ Wmsg,
                                                      const float* __restrict__ Wapp,
                                                      f16* __restrict__ wmsgf,
                                                      f16* __restrict__ wappf) {
  const int FR_MSG = NT * S_MSG * 32;  // 1536 fragment-lanes
  const int FR_APP = NT * S_APP * 32;  // 2048 fragment-lanes
  int ft = blockIdx.x * 256 + threadIdx.x;
  if (ft >= FR_MSG + FR_APP) return;
  bool is_msg = ft < FR_MSG;
  int f      = is_msg ? ft : ft - FR_MSG;
  int steps  = is_msg ? S_MSG : S_APP;
  int K      = is_msg ? KMSG : KAPP;
  const float* W = is_msg ? Wmsg : Wapp;
  f16* out = (is_msg ? wmsgf : wappf) + (size_t)f * 16;

  int lane = f & 31;
  int s = (f >> 5) % steps;
  int t = (f >> 5) / steps;
  int n  = 16 * t + (lane & 15);
  int hi = lane >> 4;  // 0 or 1
#pragma unroll
  for (int e = 0; e < 16; ++e) {
    int klocal = (e < 8 ? e : e + 8) + hi * 8;  // {0..7,16..23} or {8..15,24..31}
    int k = 32 * s + klocal;
    out[e] = (f16)W[(size_t)n * K + k];         // B[k][n] = W[n][k]
  }
}

// ---------------------------------------------------------------------------
// Kernel 2: fused gather + mean-pool (linearity: pool before GEMM).
// One wave per node. pooled[n] = (1/16)*[ sum src_h | sum e_h ]  (f16, 192)
// ---------------------------------------------------------------------------
__global__ __launch_bounds__(256) void pool_kernel(const float* __restrict__ nfeats,
                                                   const float* __restrict__ efeats,
                                                   const int*   __restrict__ src_idx,
                                                   f16* __restrict__ pooled) {
  int lane = threadIdx.x & 31;
  int node = blockIdx.x * 8 + (threadIdx.x >> 5);
  if (node >= N_NODES) return;

  float a0 = 0.f, a1 = 0.f, a2 = 0.f, a3 = 0.f, e0 = 0.f, e1 = 0.f;
  long ebase = (long)node * NNB;
#pragma unroll 4
  for (int j = 0; j < NNB; ++j) {
    int s = src_idx[ebase + j];
    float4 sv = *(const float4*)(nfeats + (long)s * DIN + lane * 4);       // 512B/wave, L2-resident
    float2 ev = *(const float2*)(efeats + (ebase + j) * DE + lane * 2);    // 256B/wave, HBM stream
    a0 += sv.x; a1 += sv.y; a2 += sv.z; a3 += sv.w;
    e0 += ev.x; e1 += ev.y;
  }
  const float sc = 1.0f / 16.0f;
  f16* row = pooled + (long)node * KMSG;
  v4h pv = { (f16)(a0 * sc), (f16)(a1 * sc), (f16)(a2 * sc), (f16)(a3 * sc) };
  *(v4h*)(row + lane * 4) = pv;                  // src part: k = 0..127
  v2h ev2 = { (f16)(e0 * sc), (f16)(e1 * sc) };
  *(v2h*)(row + DIN + lane * 2) = ev2;           // edge part: k = 128..191
}

// ---------------------------------------------------------------------------
// Kernel 3: one wave per 16-node tile.
//   GEMM1: h_neigh[16x128] = pooled[16x192] x Wmsg^T      (6x8 WMMAs)
//   LDS A2[16][256] = [ nfeats_f16 | h_neigh_f16 + b_msg ]
//   GEMM2: out = relu( A2[16x256] x Wapp^T + b_apply )    (8x8 WMMAs)
// ---------------------------------------------------------------------------
__global__ __launch_bounds__(32) void sage_main(const float* __restrict__ nfeats,
                                                const f16*   __restrict__ pooled,
                                                const f16*   __restrict__ wmsgf,
                                                const f16*   __restrict__ wappf,
                                                const float* __restrict__ bmsg,
                                                const float* __restrict__ bapp,
                                                float* __restrict__ out) {
  __shared__ f16 A2[16][KAPP];  // 8 KB

  int lane = threadIdx.x;
  int m0   = blockIdx.x * 16;
  int col  = lane & 15;
  int hi   = lane >> 4;  // 0 or 1

  // stage nfeats (f32 -> f16) into A2[:, 0:128]
#pragma unroll 4
  for (int mm = 0; mm < 16; ++mm) {
    float4 v = *(const float4*)(nfeats + (long)(m0 + mm) * DIN + lane * 4);
    v4h h = { (f16)v.x, (f16)v.y, (f16)v.z, (f16)v.w };
    *(v4h*)(&A2[mm][lane * 4]) = h;
  }

  const v8f zero = { 0.f, 0.f, 0.f, 0.f, 0.f, 0.f, 0.f, 0.f };

  // ---- GEMM1: message matmul on pooled features ----
  v8f acc[NT];
#pragma unroll
  for (int t = 0; t < NT; ++t) acc[t] = zero;

#pragma unroll
  for (int s = 0; s < S_MSG; ++s) {
    // A-fragment: row = m0+col, two 16B chunks at k = 32s+{0,16} (+8 for hi lanes)
    const f16* ap = pooled + (long)(m0 + col) * KMSG + 32 * s + hi * 8;
    v8h lo = *(const v8h*)(ap);
    v8h hh = *(const v8h*)(ap + 16);
    v16h a;
#pragma unroll
    for (int i = 0; i < 8; ++i) { a[i] = lo[i]; a[i + 8] = hh[i]; }
#pragma unroll
    for (int t = 0; t < NT; ++t) {
      v16h b = *(const v16h*)(wmsgf + ((size_t)(t * S_MSG + s) * 32 + lane) * 16);
      acc[t] = __builtin_amdgcn_wmma_f32_16x16x32_f16(false, a, false, b,
                                                      (short)0, acc[t], false, false);
    }
  }

  // h_neigh + b_msg -> f16 into A2[:, 128:256]
#pragma unroll
  for (int t = 0; t < NT; ++t) {
    int och = 16 * t + col;
    float bb = bmsg[och];
#pragma unroll
    for (int r = 0; r < 8; ++r) {
      int m = r + hi * 8;  // C/D layout: lanes<16 -> M=r, lanes>=16 -> M=r+8
      A2[m][DIN + och] = (f16)(acc[t][r] + bb);
    }
  }
  __syncthreads();  // single-wave WG: lowers to DS-counter wait

  // ---- GEMM2: apply matmul on [nfeats | h_neigh] from LDS ----
  v8f acc2[NT];
#pragma unroll
  for (int t = 0; t < NT; ++t) acc2[t] = zero;

#pragma unroll
  for (int s = 0; s < S_APP; ++s) {
    const f16* ap = &A2[col][32 * s + hi * 8];
    v8h lo = *(const v8h*)(ap);
    v8h hh = *(const v8h*)(ap + 16);
    v16h a;
#pragma unroll
    for (int i = 0; i < 8; ++i) { a[i] = lo[i]; a[i + 8] = hh[i]; }
#pragma unroll
    for (int t = 0; t < NT; ++t) {
      v16h b = *(const v16h*)(wappf + ((size_t)(t * S_APP + s) * 32 + lane) * 16);
      acc2[t] = __builtin_amdgcn_wmma_f32_16x16x32_f16(false, a, false, b,
                                                       (short)0, acc2[t], false, false);
    }
  }

  // epilogue: bias + ReLU, f32 store
#pragma unroll
  for (int t = 0; t < NT; ++t) {
    int och = 16 * t + col;
    float bb = bapp[och];
#pragma unroll
    for (int r = 0; r < 8; ++r) {
      int m = m0 + r + hi * 8;
      float v = acc2[t][r] + bb;
      out[(long)m * DOUT + och] = v > 0.f ? v : 0.f;
    }
  }
}

// ---------------------------------------------------------------------------
// Host launch.  Inputs (setup_inputs order):
//   0 nfeats[N,1,128] f32   1 efeats[E,1,64] f32   2 W_msg[128,192] f32
//   3 b_msg[128] f32        4 W_apply[128,256] f32 5 b_apply[128] f32
//   6 src_idx[E] i32        7 dst_idx[E] i32 (unused: dst = repeat(arange,16))
// Workspace: pooled f16 N*192 (18.3MB) + weight fragments (~112KB).
// ---------------------------------------------------------------------------
extern "C" void kernel_launch(void* const* d_in, const int* in_sizes, int n_in,
                              void* d_out, int out_size, void* d_ws, size_t ws_size,
                              hipStream_t stream) {
  const float* nfeats = (const float*)d_in[0];
  const float* efeats = (const float*)d_in[1];
  const float* Wmsg   = (const float*)d_in[2];
  const float* bmsg   = (const float*)d_in[3];
  const float* Wapp   = (const float*)d_in[4];
  const float* bapp   = (const float*)d_in[5];
  const int*   src    = (const int*)d_in[6];
  float* out = (float*)d_out;

  f16* pooled = (f16*)d_ws;
  f16* wmsgf  = pooled + (size_t)N_NODES * KMSG;
  f16* wappf  = wmsgf + (size_t)NT * S_MSG * 32 * 16;

  const int total_frag = NT * S_MSG * 32 + NT * S_APP * 32;  // 3584
  repack_weights<<<(total_frag + 255) / 256, 256, 0, stream>>>(Wmsg, Wapp, wmsgf, wappf);
  pool_kernel<<<(N_NODES + 7) / 8, 256, 0, stream>>>(nfeats, efeats, src, pooled);
  sage_main<<<N_NODES / 16, 32, 0, stream>>>(nfeats, pooled, wmsgf, wappf, bmsg, bapp, out);
}
